// MyEncoder_18116172054918
// MI455X (gfx1250) — matrix operand order repaired and checked by reading, not compiled, not run on hardware
//
#include <hip/hip_runtime.h>
#include <hip/hip_bf16.h>
#include <math.h>

typedef __attribute__((ext_vector_type(16))) _Float16 v16h;
typedef __attribute__((ext_vector_type(8)))  _Float16 v8h;
typedef __attribute__((ext_vector_type(8)))  float    v8f;

#define EMB   1024
#define SEQ   2048
#define BATCH 2
#define HEADS 16
#define HDIM  64
#define ROWS  (BATCH*SEQ)   // 4096
#define SCALE 0.125f        // 1/sqrt(64)

// ---------------------------------------------------------------------------
// WMMA fragment loaders (wave32 layouts, CDNA5 ISA §7.12.2)
__device__ inline v16h load_a16(const _Float16* __restrict__ p0, int ld, int lane) {
    int row  = lane & 15;
    int kOff = (lane & 16) ? 8 : 0;
    const _Float16* p = p0 + row * ld + kOff;
    v8h lo = *(const v8h*)(p);
    v8h hi = *(const v8h*)(p + 16);
    v16h a;
#pragma unroll
    for (int i = 0; i < 8; ++i) { a[i] = lo[i]; a[8 + i] = hi[i]; }
    return a;
}

__device__ inline v16h load_b16(const _Float16* __restrict__ p0, int ld, int lane) {
    int n     = lane & 15;
    int kBase = (lane & 16) ? 8 : 0;
    const _Float16* p = p0 + n * ld + kBase;
    v8h lo = *(const v8h*)(p);
    v8h hi = *(const v8h*)(p + 16);
    v16h b;
#pragma unroll
    for (int i = 0; i < 8; ++i) { b[i] = lo[i]; b[8 + i] = hi[i]; }
    return b;
}

__device__ inline v8f wmma_f16(v16h a, v16h b, v8f c) {
    return __builtin_amdgcn_wmma_f32_16x16x32_f16(false, a, false, b, (short)0, c, false, false);
}

// ---------------------------------------------------------------------------
// LayerNorm (unbiased std, torch-style) + cast to f16. One block per row.
__global__ __launch_bounds__(256) void ln_cast_kernel(const float* __restrict__ x,
                                                      _Float16* __restrict__ xn) {
    __shared__ float s_sum[256], s_sq[256];
    int row = blockIdx.x;
    int tid = threadIdx.x;
    const float* xr = x + (size_t)row * EMB;
    float v[4];
    float sum = 0.f, sq = 0.f;
#pragma unroll
    for (int i = 0; i < 4; ++i) { v[i] = xr[tid + i * 256]; sum += v[i]; sq += v[i] * v[i]; }
    s_sum[tid] = sum; s_sq[tid] = sq;
    __syncthreads();
    for (int off = 128; off > 0; off >>= 1) {
        if (tid < off) { s_sum[tid] += s_sum[tid + off]; s_sq[tid] += s_sq[tid + off]; }
        __syncthreads();
    }
    float mean = s_sum[0] * (1.0f / EMB);
    float var  = (s_sq[0] - (float)EMB * mean * mean) * (1.0f / (EMB - 1));
    float sd   = sqrtf(fmaxf(var, 0.f));
    float inv  = 1.0f / (sd + 1e-6f);
    _Float16* xo = xn + (size_t)row * EMB;
#pragma unroll
    for (int i = 0; i < 4; ++i) xo[tid + i * 256] = (_Float16)((v[i] - mean) * inv);
}

__global__ __launch_bounds__(256) void cast_f32_f16(const float* __restrict__ src,
                                                    _Float16* __restrict__ dst, int n) {
    int i = blockIdx.x * blockDim.x + threadIdx.x;
    if (i < n) dst[i] = (_Float16)src[i];
}

// ---------------------------------------------------------------------------
// GEMM: Y[M=4096, N=1024] = A[f16] @ W^T[f16] + bias, f32 accumulate via WMMA.
// Wave computes a 16x64 tile (4 accumulators share one A-frag). Block = 8 waves
// = 64x128 tile. Ping-pong double buffering (2x-unrolled K-loop, two fixed
// fragment buffer sets) so loads overlap WMMA with zero register copies.
// mode 0: f16 out [B,H,S,Dh]; mode 1: f16 out [B,H,Dh,S] (V transposed);
// mode 2: f32 out [B,S,E].
__global__ __launch_bounds__(256) void gemm_kernel(const _Float16* __restrict__ A,
                                                   const _Float16* __restrict__ W,
                                                   const float* __restrict__ bias,
                                                   void* __restrict__ out, int mode) {
    int lane = threadIdx.x & 31;
    int wave = threadIdx.x >> 5;
    int wm = wave & 3;          // 4 waves along M
    int wn = wave >> 2;         // 2 waves along N
    int m0 = blockIdx.x * 64 + wm * 16;
    int n0 = blockIdx.y * 128 + wn * 64;

    const _Float16* Ab = A + (size_t)m0 * EMB;
    const _Float16* Wb[4];
#pragma unroll
    for (int t = 0; t < 4; ++t) Wb[t] = W + (size_t)(n0 + t * 16) * EMB;

    v8f c[4] = {};
    // Ping-pong buffers; prologue fills buffer 0 with k = 0.
    v16h a0 = load_a16(Ab, EMB, lane), a1;
    v16h b0[4], b1[4];
#pragma unroll
    for (int t = 0; t < 4; ++t) b0[t] = load_b16(Wb[t], EMB, lane);

    int k = 0;
    for (; k < EMB - 64; k += 64) {
        // fill buffer 1 with k+32 while consuming buffer 0 (k)
        __builtin_prefetch(Ab + (lane & 15) * EMB + k + 128, 0, 3);  // global_prefetch_b8, WGP scope
        a1 = load_a16(Ab + k + 32, EMB, lane);
#pragma unroll
        for (int t = 0; t < 4; ++t) b1[t] = load_b16(Wb[t] + k + 32, EMB, lane);
#pragma unroll
        for (int t = 0; t < 4; ++t) c[t] = wmma_f16(a0, b0[t], c[t]);
        // fill buffer 0 with k+64 while consuming buffer 1 (k+32)
        a0 = load_a16(Ab + k + 64, EMB, lane);
#pragma unroll
        for (int t = 0; t < 4; ++t) b0[t] = load_b16(Wb[t] + k + 64, EMB, lane);
#pragma unroll
        for (int t = 0; t < 4; ++t) c[t] = wmma_f16(a1, b1[t], c[t]);
    }
    // Tail: k == EMB-64; consume k and k+32.
    a1 = load_a16(Ab + k + 32, EMB, lane);
#pragma unroll
    for (int t = 0; t < 4; ++t) b1[t] = load_b16(Wb[t] + k + 32, EMB, lane);
#pragma unroll
    for (int t = 0; t < 4; ++t) c[t] = wmma_f16(a0, b0[t], c[t]);
#pragma unroll
    for (int t = 0; t < 4; ++t) c[t] = wmma_f16(a1, b1[t], c[t]);

    int N     = lane & 15;
    int mBase = (lane & 16) ? 8 : 0;   // C/D layout: VGPR r -> M = mBase + r
#pragma unroll
    for (int t = 0; t < 4; ++t) {
        int n = n0 + t * 16 + N;
        float bv = bias[n];
        int h = n >> 6, d = n & 63;
#pragma unroll
        for (int r = 0; r < 8; ++r) {
            int m = m0 + mBase + r;
            float val = c[t][r] + bv;
            if (mode == 2) {
                ((float*)out)[(size_t)m * EMB + n] = val;
            } else {
                int b_ = m >> 11, s = m & (SEQ - 1);
                size_t idx = (mode == 0)
                    ? ((size_t)(b_ * HEADS + h) * SEQ + s) * HDIM + d
                    : ((size_t)(b_ * HEADS + h) * HDIM + d) * SEQ + s;
                ((_Float16*)out)[idx] = (_Float16)val;
            }
        }
    }
}

// ---------------------------------------------------------------------------
// Attention: one block = 16 query rows of one (b,h). Scores via WMMA -> LDS
// (f16), full-row softmax stats, exp in place, PV via WMMA (A from LDS,
// B from transposed V), 1/rowsum folded into epilogue. Hout f16 [B,S,E].
// Both WMMA loops use ping-pong double buffering (no rotation copies).
__global__ __launch_bounds__(256) void attn_kernel(const _Float16* __restrict__ Q,
                                                   const _Float16* __restrict__ K,
                                                   const _Float16* __restrict__ Vt,
                                                   _Float16* __restrict__ Hout) {
    extern __shared__ char smem[];
    _Float16* sP      = (_Float16*)smem;                       // [16][SEQ] 64KB
    float*    sRed    = (float*)(smem + 16 * SEQ * 2);         // [16][16]
    float*    sRowMax = sRed + 256;                            // [16]
    float*    sRowSum = sRowMax + 16;                          // [16]
    float*    sPart   = sRowSum + 16;                          // [4][16][16]

    int lane = threadIdx.x & 31;
    int wave = threadIdx.x >> 5;
    int bh = blockIdx.y;            // b*HEADS + h
    int s0 = blockIdx.x * 16;

    const _Float16* Qb = Q  + ((size_t)bh * SEQ + s0) * HDIM;
    const _Float16* Kb = K  + (size_t)bh * SEQ * HDIM;
    const _Float16* Vb = Vt + (size_t)bh * HDIM * SEQ;

    int N     = lane & 15;
    int mBase = (lane & 16) ? 8 : 0;

    // Phase 1: scores = Q K^T * scale. Q fragments loop-invariant; K fragment
    // pairs ping-ponged across a 2x-unrolled tile loop (8 iters x 2 tiles).
    v16h aq0 = load_a16(Qb + 0,  HDIM, lane);
    v16h aq1 = load_a16(Qb + 32, HDIM, lane);
    {
        v16h pa0 = load_b16(Kb + (size_t)wave * 16 * HDIM + 0,  HDIM, lane);
        v16h pa1 = load_b16(Kb + (size_t)wave * 16 * HDIM + 32, HDIM, lane);
        v16h pb0, pb1;
        for (int i = 0; i < 8; ++i) {
            int tc = wave + 16 * i;          // buffer-a tile
            int tb = tc + 8;                 // buffer-b tile (always < 128)
            pb0 = load_b16(Kb + (size_t)tb * 16 * HDIM + 0,  HDIM, lane);
            pb1 = load_b16(Kb + (size_t)tb * 16 * HDIM + 32, HDIM, lane);
            v8f c = {};
            c = wmma_f16(aq0, pa0, c);
            c = wmma_f16(aq1, pa1, c);
#pragma unroll
            for (int r = 0; r < 8; ++r)
                sP[(size_t)(mBase + r) * SEQ + tc * 16 + N] = (_Float16)(c[r] * SCALE);
            int tn = (i < 7) ? tc + 16 : tc;  // clamp: last iter reloads
            pa0 = load_b16(Kb + (size_t)tn * 16 * HDIM + 0,  HDIM, lane);
            pa1 = load_b16(Kb + (size_t)tn * 16 * HDIM + 32, HDIM, lane);
            v8f d = {};
            d = wmma_f16(aq0, pb0, d);
            d = wmma_f16(aq1, pb1, d);
#pragma unroll
            for (int r = 0; r < 8; ++r)
                sP[(size_t)(mBase + r) * SEQ + tb * 16 + N] = (_Float16)(d[r] * SCALE);
        }
    }
    __syncthreads();

    // Phase 2: softmax stats (16 threads per row)
    int row = threadIdx.x >> 4;
    int sub = threadIdx.x & 15;
    float mx = -3.4e38f;
    for (int tc = sub; tc < SEQ; tc += 16) mx = fmaxf(mx, (float)sP[(size_t)row * SEQ + tc]);
    sRed[row * 16 + sub] = mx;
    __syncthreads();
    if (sub == 0) {
        float m2 = sRed[row * 16];
        for (int i = 1; i < 16; ++i) m2 = fmaxf(m2, sRed[row * 16 + i]);
        sRowMax[row] = m2;
    }
    __syncthreads();
    float rmax = sRowMax[row];
    float sum = 0.f;
    for (int tc = sub; tc < SEQ; tc += 16) {
        float e = __expf((float)sP[(size_t)row * SEQ + tc] - rmax);
        sP[(size_t)row * SEQ + tc] = (_Float16)e;   // unnormalized prob, in place
        sum += e;
    }
    __syncthreads();
    sRed[row * 16 + sub] = sum;
    __syncthreads();
    if (sub == 0) {
        float s2 = 0.f;
        for (int i = 0; i < 16; ++i) s2 += sRed[row * 16 + i];
        sRowSum[row] = s2;
    }
    __syncthreads();

    // Phase 3: O = P @ V. 8 waves = 4 d-tiles x 2 K-halves of the t dimension.
    // A (probs) from LDS, B (V^T) from global; ping-pong double buffered.
    int dt = wave & 3;
    int kh = wave >> 2;
    int n0 = dt * 16;
    v8f c = {};
    {
        const int tBeg = kh * (SEQ / 2), tEnd = (kh + 1) * (SEQ / 2);
        const _Float16* Pp = (const _Float16*)sP;
        const _Float16* Vp = Vb + (size_t)n0 * SEQ;
        v16h a0 = load_a16(Pp + tBeg, SEQ, lane);
        v16h b0 = load_b16(Vp + tBeg, SEQ, lane);
        v16h a1, b1;
        for (int t0 = tBeg; t0 < tEnd; t0 += 64) {
            a1 = load_a16(Pp + t0 + 32, SEQ, lane);   // t0+32 always < tEnd
            b1 = load_b16(Vp + t0 + 32, SEQ, lane);
            c = wmma_f16(a0, b0, c);
            int tn = (t0 + 64 < tEnd) ? t0 + 64 : t0; // clamp: last iter reloads
            a0 = load_a16(Pp + tn, SEQ, lane);
            b0 = load_b16(Vp + tn, SEQ, lane);
            c = wmma_f16(a1, b1, c);
        }
    }
    if (kh == 1) {
#pragma unroll
        for (int r = 0; r < 8; ++r) sPart[(dt * 16 + (mBase + r)) * 16 + N] = c[r];
    }
    __syncthreads();
    if (kh == 0) {
        int b_ = bh >> 4, h = bh & 15;
#pragma unroll
        for (int r = 0; r < 8; ++r) {
            int srow = mBase + r;
            float val = (c[r] + sPart[(dt * 16 + srow) * 16 + N]) / sRowSum[srow];
            Hout[((size_t)b_ * SEQ + s0 + srow) * EMB + h * HDIM + n0 + N] = (_Float16)val;
        }
    }
}

// ---------------------------------------------------------------------------
extern "C" void kernel_launch(void* const* d_in, const int* in_sizes, int n_in,
                              void* d_out, int out_size, void* d_ws, size_t ws_size,
                              hipStream_t stream) {
    const float* x  = (const float*)d_in[0];
    const float* wQ = (const float*)d_in[1];
    const float* bQ = (const float*)d_in[2];
    const float* wK = (const float*)d_in[3];
    const float* bK = (const float*)d_in[4];
    const float* wV = (const float*)d_in[5];
    const float* bV = (const float*)d_in[6];
    const float* wO = (const float*)d_in[7];
    const float* bO = (const float*)d_in[8];
    float* out = (float*)d_out;

    const size_t ACT = (size_t)ROWS * EMB * sizeof(_Float16);  // 8 MB
    const size_t WGT = (size_t)EMB * EMB * sizeof(_Float16);   // 2 MB
    char* ws = (char*)d_ws;
    _Float16* xn   = (_Float16*)(ws);                  // 8 MB
    _Float16* wQh  = (_Float16*)(ws + ACT);            // 2 MB
    _Float16* wKh  = (_Float16*)(ws + ACT + WGT);
    _Float16* wVh  = (_Float16*)(ws + ACT + 2 * WGT);
    _Float16* wOh  = (_Float16*)(ws + ACT + 3 * WGT);
    _Float16* Qh   = (_Float16*)(ws + ACT + 4 * WGT);          // [B,H,S,Dh]
    _Float16* Kh   = (_Float16*)(ws + 2 * ACT + 4 * WGT);      // [B,H,S,Dh]
    _Float16* Vth  = (_Float16*)(ws + 3 * ACT + 4 * WGT);      // [B,H,Dh,S]
    _Float16* Hth  = (_Float16*)(ws + 4 * ACT + 4 * WGT);      // [B,S,E]

    ln_cast_kernel<<<ROWS, 256, 0, stream>>>(x, xn);
    cast_f32_f16<<<(EMB * EMB) / 256, 256, 0, stream>>>(wQ, wQh, EMB * EMB);
    cast_f32_f16<<<(EMB * EMB) / 256, 256, 0, stream>>>(wK, wKh, EMB * EMB);
    cast_f32_f16<<<(EMB * EMB) / 256, 256, 0, stream>>>(wV, wVh, EMB * EMB);
    cast_f32_f16<<<(EMB * EMB) / 256, 256, 0, stream>>>(wO, wOh, EMB * EMB);

    dim3 ggrid(ROWS / 64, EMB / 128);
    gemm_kernel<<<ggrid, 256, 0, stream>>>(xn, wQh, bQ, (void*)Qh,  0);
    gemm_kernel<<<ggrid, 256, 0, stream>>>(xn, wKh, bK, (void*)Kh,  0);
    gemm_kernel<<<ggrid, 256, 0, stream>>>(xn, wVh, bV, (void*)Vth, 1);

    size_t smem = (size_t)16 * SEQ * 2          // sP
                + 256 * 4 + 16 * 4 + 16 * 4     // sRed, sRowMax, sRowSum
                + 4 * 16 * 16 * 4;              // sPart
    (void)hipFuncSetAttribute((const void*)attn_kernel,
                              hipFuncAttributeMaxDynamicSharedMemorySize, (int)smem);
    attn_kernel<<<dim3(SEQ / 16, BATCH * HEADS), 256, smem, stream>>>(Qh, Kh, Vth, Hth);

    gemm_kernel<<<ggrid, 256, 0, stream>>>(Hth, wOh, bO, (void*)out, 2);
}
